// TemporalCorrespondanceModule_28260884807990
// MI455X (gfx1250) — matrix operand order repaired and checked by reading, not compiled, not run on hardware
//
#include <hip/hip_runtime.h>
#include <hip/hip_bf16.h>

typedef __attribute__((ext_vector_type(16))) _Float16     v16h;
typedef __attribute__((ext_vector_type(8)))  float        v8f;
typedef __attribute__((ext_vector_type(4)))  unsigned int u32x4;

#define HDIM   128
#define WDIM   128
#define FMC    64
#define NKP    10
#define WIN    13
#define NDISP  169
#define NDISPP 192

// async global->LDS staging (gfx1250 GLOBAL_LOAD_ASYNC_TO_LDS_B128 + ASYNCcnt)
#ifndef USE_ASYNC_LDS
#define USE_ASYNC_LDS 1
#endif
#define LDS_AS __attribute__((address_space(3)))

union HFrag { u32x4 u[2]; v16h v; };

// ---------------------------------------------------------------------------
// Implicit-GEMM conv (NHWC f16 activations, packed-fragment f16 weights).
// Each wave computes a 32-pixel x 64-outchannel tile: 2 A-frags x 4 B-frags
// -> 8 v_wmma_f32_16x16x32_f16 per K-step. Up to 8 waves per block share the
// 64x32 weight tile, staged once per (tap, K-chunk) into LDS via the async
// global->LDS engine. Supports zero padding, fused bias+ReLU, nearest-2x-
// upsampled input, channel concat (C1 multiple of 32), f16/f32 outputs.
// ---------------------------------------------------------------------------
__global__ void __launch_bounds__(256) conv_wmma_k(
    const _Float16* __restrict__ inA, const _Float16* __restrict__ inB, int C1,
    const _Float16* __restrict__ pw, const float* __restrict__ bias, int relu,
    _Float16* __restrict__ outH, float* __restrict__ outF_nhwc,
    float* __restrict__ outF_nchw,
    int H_, int W_, int CinP, int CoutP, int CoutReal,
    int KH, int KW, int pad, int up)
{
  __shared__ alignas(16) _Float16 lw[64 * 32];   // one 64-co x 32-k B tile

  const int tid    = threadIdx.x;
  const int nthr   = blockDim.x;
  const int nwaves = nthr >> 5;
  const int wave   = tid >> 5;
  const int lane   = tid & 31;
  const int n      = lane & 15;   // B/C/D column (output channel in subtile)
  const int half   = lane >> 4;   // K-half selector
  const int b      = blockIdx.z;
  const int co0    = blockIdx.y << 6;
  const int HW     = H_ * W_;
  const int pm0    = (blockIdx.x * nwaves + wave) << 5;  // 32-pixel tile base
  const int KQ     = CinP >> 5;
  const int Hs     = up ? (H_ >> 1) : H_;
  const int Ws     = up ? (W_ >> 1) : W_;

  // two pixel rows per lane (M subtiles 0..15 and 16..31)
  int yr[2], xr[2];
  {
    int pA = pm0 + n;       yr[0] = pA / W_; xr[0] = pA - yr[0] * W_;
    int pB = pm0 + 16 + n;  yr[1] = pB / W_; xr[1] = pB - yr[1] * W_;
  }

  v8f acc[2][4];
  {
    v8f z = {0.f, 0.f, 0.f, 0.f, 0.f, 0.f, 0.f, 0.f};
    for (int s = 0; s < 2; ++s)
      for (int j = 0; j < 4; ++j) acc[s][j] = z;
  }
  const u32x4 z4 = {0u, 0u, 0u, 0u};

#if USE_ASYNC_LDS
  // 32-bit wave-relative LDS base address of lw (AS3 pointers are LDS offsets)
  const unsigned lwBase = (unsigned)(unsigned long)((LDS_AS char*)lw);
#endif

  const int NT = KH * KW;
  for (int t = 0; t < NT; ++t) {
    const int ky = t / KW - pad;
    const int kx = t - (t / KW) * KW - pad;
    long pixOff[2]; bool valid[2];
    for (int s = 0; s < 2; ++s) {
      int yy = yr[s] + ky, xx = xr[s] + kx;
      valid[s] = (yy >= 0) & (yy < H_) & (xx >= 0) & (xx < W_);
      int ys = up ? (yy >> 1) : yy;
      int xs = up ? (xx >> 1) : xx;
      pixOff[s] = (long)b * Hs * Ws + (long)ys * Ws + xs;
    }

    for (int q = 0; q < KQ; ++q) {
      const int c0 = q << 5;
      const _Float16* src; int CA, cl;
      if (c0 < C1) { src = inA; CA = C1;        cl = c0;      }
      else         { src = inB; CA = CinP - C1; cl = c0 - C1; }

      // ---- stage 64x32 weight tile into LDS (256 x b128) ----
      const u32x4* wsrc =
          (const u32x4*)(pw + (((long)t * KQ + q) * CoutP + co0) * 32);
      __syncthreads();
#if USE_ASYNC_LDS
      for (int idx = tid; idx < 256; idx += nthr) {
        unsigned ldsAddr = lwBase + (unsigned)idx * 16u;
        asm volatile("global_load_async_to_lds_b128 %0, %1, off"
                     :: "v"(ldsAddr), "v"((const void*)(wsrc + idx))
                     : "memory");
      }
      asm volatile("s_wait_asynccnt 0x0" ::: "memory");
#else
      for (int idx = tid; idx < 256; idx += nthr)
        *((u32x4*)lw + idx) = wsrc[idx];
#endif
      // prefetch next tile's weights (global_prefetch_b8)
      if (t * KQ + q + 1 < NT * KQ)
        __builtin_prefetch(
            (const void*)(pw + (((long)(t * KQ + q) + 1) * CoutP + co0) * 32),
            0, 1);
      __syncthreads();

      // ---- A fragments: lane half h holds K {h*8..+7} and {16+h*8..+7} ----
      HFrag a[2];
      for (int s = 0; s < 2; ++s) {
        if (valid[s]) {
          const _Float16* p = src + pixOff[s] * CA + cl + half * 8;
          a[s].u[0] = *(const u32x4*)(p);
          a[s].u[1] = *(const u32x4*)(p + 16);
        } else { a[s].u[0] = z4; a[s].u[1] = z4; }
      }

      // ---- 4 B fragments from LDS, 8 WMMAs ----
      for (int ns = 0; ns < 4; ++ns) {
        HFrag w;
        const _Float16* lp = lw + ((ns * 16 + n) * 32) + half * 8;
        w.u[0] = *(const u32x4*)(lp);
        w.u[1] = *(const u32x4*)(lp + 16);
        acc[0][ns] = __builtin_amdgcn_wmma_f32_16x16x32_f16(
            false, a[0].v, false, w.v, (short)0, acc[0][ns], false, false);
        acc[1][ns] = __builtin_amdgcn_wmma_f32_16x16x32_f16(
            false, a[1].v, false, w.v, (short)0, acc[1][ns], false, false);
      }
    }
  }

  // ---- epilogue: bias + ReLU + stores ----
  for (int ns = 0; ns < 4; ++ns) {
    const int chan = co0 + ns * 16 + n;
    if (chan < CoutReal) {
      const float bv = bias ? bias[chan] : 0.f;
      for (int s = 0; s < 2; ++s) {
        for (int r = 0; r < 8; ++r) {
          const long pmo = (long)pm0 + s * 16 + r + half * 8;
          float v = acc[s][ns][r] + bv;
          if (relu) v = fmaxf(v, 0.f);
          if (outH)      outH[((long)b * HW + pmo) * CoutReal + chan] = (_Float16)v;
          if (outF_nhwc) outF_nhwc[((long)b * HW + pmo) * CoutReal + chan] = v;
          if (outF_nchw) outF_nchw[((long)b * CoutReal + chan) * HW + pmo] = v;
        }
      }
    }
  }
}

// ---------------------------------------------------------------------------
// Pack frame f32 NCHW(3) -> f16 NHWC(32, zero padded)
// ---------------------------------------------------------------------------
__global__ void pack_frame_k(const float* __restrict__ src,
                             _Float16* __restrict__ dst, int B_, int HW) {
  long i = (long)blockIdx.x * blockDim.x + threadIdx.x;
  long total = (long)B_ * HW * 32;
  if (i >= total) return;
  int c = (int)(i & 31);
  long pixg = i >> 5;
  long b = pixg / HW;
  long pix = pixg - b * HW;
  float v = (c < 3) ? src[(b * 3 + c) * HW + pix] : 0.f;
  dst[i] = (_Float16)v;
}

// ---------------------------------------------------------------------------
// Pack weights f32 OIHW -> f16 B-fragment layout [tap][q][co][k(0..31)]
// ---------------------------------------------------------------------------
__global__ void pack_w_k(const float* __restrict__ w, _Float16* __restrict__ pw,
                         int CinReal, int CinP, int CoutReal, int CoutP, int KHW) {
  long i = (long)blockIdx.x * blockDim.x + threadIdx.x;
  int KQ = CinP >> 5;
  long total = (long)KHW * KQ * CoutP * 32;
  if (i >= total) return;
  int kk = (int)(i & 31);
  long r = i >> 5;
  int co = (int)(r % CoutP); r /= CoutP;
  int q  = (int)(r % KQ);
  int t  = (int)(r / KQ);
  int ci = q * 32 + kk;
  float v = 0.f;
  if (ci < CinReal && co < CoutReal)
    v = w[((long)co * CinReal + ci) * KHW + t];
  pw[i] = (_Float16)v;
}

// ---------------------------------------------------------------------------
// 2x2 average pool, NHWC f16
// ---------------------------------------------------------------------------
__global__ void pool2_k(const _Float16* __restrict__ in,
                        _Float16* __restrict__ out, int B_, int ho, int wo, int C) {
  long i = (long)blockIdx.x * blockDim.x + threadIdx.x;
  long total = (long)B_ * ho * wo * C;
  if (i >= total) return;
  int c = (int)(i % C); long r = i / C;
  int xo = (int)(r % wo); r /= wo;
  int yo = (int)(r % ho);
  int b  = (int)(r / ho);
  int wi = wo * 2;
  const _Float16* base = in + ((long)b * (ho * 2)) * wi * C + c;
  float s = (float)base[((long)(2 * yo)     * wi + 2 * xo)     * C]
          + (float)base[((long)(2 * yo)     * wi + 2 * xo + 1) * C]
          + (float)base[((long)(2 * yo + 1) * wi + 2 * xo)     * C]
          + (float)base[((long)(2 * yo + 1) * wi + 2 * xo + 1) * C];
  out[i] = (_Float16)(0.25f * s);
}

// ---------------------------------------------------------------------------
// Feature-map normalization: per-(b,c) spatial mean, then per-pixel L2 over c
// ---------------------------------------------------------------------------
__global__ void fm_mean_k(const float* __restrict__ fm, float* __restrict__ means,
                          int HW, int C) {
  __shared__ float red[256];
  int bc = blockIdx.x;
  int b = bc / C, c = bc - b * C;
  int t = threadIdx.x;
  float s = 0.f;
  for (int i = t; i < HW; i += 256) s += fm[((long)b * HW + i) * C + c];
  red[t] = s; __syncthreads();
  for (int st = 128; st > 0; st >>= 1) {
    if (t < st) red[t] += red[t + st];
    __syncthreads();
  }
  if (t == 0) means[bc] = red[0] / (float)HW;
}

__global__ void fm_norm_k(float* __restrict__ fm, const float* __restrict__ means,
                          int B_, int HW, int C) {
  long i = (long)blockIdx.x * blockDim.x + threadIdx.x;
  if (i >= (long)B_ * HW) return;
  int b = (int)(i / HW);
  float* p = fm + i * C;
  const float* mb = means + (long)b * C;
  float ss = 0.f;
  for (int c = 0; c < C; ++c) { float v = p[c] - mb[c]; ss += v * v; }
  float rs = rsqrtf(fmaxf(ss, 1e-20f));
  for (int c = 0; c < C; ++c) p[c] = (p[c] - mb[c]) * rs;
}

// ---------------------------------------------------------------------------
// Cost volume: 169 displacements, zero-padded fm0, per-pixel dot over 64 ch.
// ---------------------------------------------------------------------------
__global__ void costvol_k(const float* __restrict__ fm0,
                          const float* __restrict__ fm1,
                          float* __restrict__ corrF,
                          _Float16* __restrict__ corrH, int H_, int W_) {
  __shared__ float f1[FMC];
  int x = blockIdx.x, y = blockIdx.y, b = blockIdx.z, t = threadIdx.x;
  int HW = H_ * W_;
  long pix = (long)y * W_ + x;
  if (t < FMC) f1[t] = fm1[((long)b * HW + pix) * FMC + t];
  __syncthreads();
  if (t < NDISP) {
    int dy = t / WIN - 6, dx = t - (t / WIN) * WIN - 6;
    int yy = y + dy, xx = x + dx;
    float dot = 0.f;
    if (yy >= 0 && yy < H_ && xx >= 0 && xx < W_) {
      const float* p0 = fm0 + ((long)b * HW + (long)yy * W_ + xx) * FMC;
      for (int c = 0; c < FMC; ++c) dot += f1[c] * p0[c];
    }
    dot *= (1.f / (float)FMC);
    corrF[((long)b * HW + pix) * NDISP + t]  = dot;
    corrH[((long)b * HW + pix) * NDISPP + t] = (_Float16)dot;
  } else if (t < NDISPP) {
    corrH[((long)b * HW + pix) * NDISPP + t] = (_Float16)0.f;
  }
}

// ---------------------------------------------------------------------------
// Per-pixel softmax over 169 displacements (T=0.1) + edge-padded shifted
// aggregation of pv0 (10 ch) and pj0 (40 ch).
// ---------------------------------------------------------------------------
__global__ void smax_agg_k(const float* __restrict__ corrF,
                           const float* __restrict__ pv0,
                           const float* __restrict__ pj0,
                           float* __restrict__ pv1, float* __restrict__ pj1,
                           int H_, int W_) {
  __shared__ float prob[NDISP];
  __shared__ float red[256];
  int x = blockIdx.x, y = blockIdx.y, b = blockIdx.z, t = threadIdx.x;
  int HW = H_ * W_;
  long pix = (long)y * W_ + x;

  float v = (t < NDISP) ? corrF[((long)b * HW + pix) * NDISP + t] : -1e30f;
  red[t] = v; __syncthreads();
  for (int s = 128; s > 0; s >>= 1) {
    if (t < s) red[t] = fmaxf(red[t], red[t + s]);
    __syncthreads();
  }
  float mx = red[0]; __syncthreads();
  float e = (t < NDISP) ? __expf((v - mx) * 10.f) : 0.f;
  red[t] = e; __syncthreads();
  for (int s = 128; s > 0; s >>= 1) {
    if (t < s) red[t] += red[t + s];
    __syncthreads();
  }
  float inv = 1.f / red[0];
  if (t < NDISP) prob[t] = e * inv;
  __syncthreads();

  if (t < NKP + 4 * NKP) {
    const float* src = (t < NKP)
        ? pv0 + ((long)b * NKP + t) * HW
        : pj0 + ((long)b * 4 * NKP + (t - NKP)) * HW;
    float acc = 0.f;
    for (int d = 0; d < NDISP; ++d) {
      int dy = d / WIN - 6, dx = d - (d / WIN) * WIN - 6;
      int yy = min(max(y + dy, 0), H_ - 1);
      int xx = min(max(x + dx, 0), W_ - 1);
      acc += prob[d] * src[(long)yy * W_ + xx];
    }
    if (t < NKP) pv1[((long)b * NKP + t) * HW + pix] = acc;
    else         pj1[((long)b * 4 * NKP + (t - NKP)) * HW + pix] = acc;
  }
}

// ---------------------------------------------------------------------------
// Heatmap softmax over HW (T=0.1) + soft-argmax on [-1,1] grid.
// ---------------------------------------------------------------------------
__global__ void heat_value_k(const float* __restrict__ pv1,
                             float* __restrict__ heat, float* __restrict__ value,
                             int H_, int W_) {
  const int kp = blockIdx.x, b = blockIdx.y, t = threadIdx.x;
  const int HW = H_ * W_;
  const float* src = pv1 + ((long)b * NKP + kp) * HW;
  float* dst = heat + ((long)b * NKP + kp) * HW;
  __shared__ float red[256];

  float mx = -1e30f;
  for (int i = t; i < HW; i += 256) mx = fmaxf(mx, src[i]);
  red[t] = mx; __syncthreads();
  for (int s = 128; s > 0; s >>= 1) {
    if (t < s) red[t] = fmaxf(red[t], red[t + s]);
    __syncthreads();
  }
  const float M = red[0]; __syncthreads();

  float s0 = 0.f, sx = 0.f, sy = 0.f;
  const float gxs = 2.f / (float)(W_ - 1), gys = 2.f / (float)(H_ - 1);
  for (int i = t; i < HW; i += 256) {
    float e = __expf((src[i] - M) * 10.f);
    dst[i] = e;
    int yy = i / W_, xx = i - yy * W_;
    s0 += e; sx += e * (gxs * xx - 1.f); sy += e * (gys * yy - 1.f);
  }
  red[t] = s0; __syncthreads();
  for (int s = 128; s > 0; s >>= 1) { if (t < s) red[t] += red[t + s]; __syncthreads(); }
  const float S = red[0]; __syncthreads();
  red[t] = sx; __syncthreads();
  for (int s = 128; s > 0; s >>= 1) { if (t < s) red[t] += red[t + s]; __syncthreads(); }
  const float SX = red[0]; __syncthreads();
  red[t] = sy; __syncthreads();
  for (int s = 128; s > 0; s >>= 1) { if (t < s) red[t] += red[t + s]; __syncthreads(); }
  const float SY = red[0];

  const float inv = 1.f / S;
  for (int i = t; i < HW; i += 256) dst[i] *= inv;
  if (t == 0) {
    value[((long)b * NKP + kp) * 2 + 0] = SX * inv;
    value[((long)b * NKP + kp) * 2 + 1] = SY * inv;
  }
}

__global__ void zero_f32_k(float* __restrict__ p, long n) {
  long i = (long)blockIdx.x * blockDim.x + threadIdx.x;
  if (i < n) p[i] = 0.f;
}

// ---------------------------------------------------------------------------
// j1 = jm @ j0 per (b,kp,pixel); pred_jac1 store; jac1 = sum heat*j1.
// ---------------------------------------------------------------------------
__global__ void jac_k(const float* __restrict__ jm, const float* __restrict__ pj1,
                      const float* __restrict__ heat,
                      float* __restrict__ pred_jac1, float* __restrict__ jac1,
                      int HW) {
  __shared__ float red[256];
  const int t = threadIdx.x;
  const int pix = blockIdx.x * 256 + t;
  const int kp = blockIdx.y, b = blockIdx.z;
  const long base = ((long)b * 4 * NKP + kp * 4) * HW + pix;

  float ja = jm[base], jb_ = jm[base + HW], jc = jm[base + 2L * HW], jd = jm[base + 3L * HW];
  float e = pj1[base], f = pj1[base + HW], g = pj1[base + 2L * HW], h = pj1[base + 3L * HW];
  float j00 = ja * e + jb_ * g;
  float j01 = ja * f + jb_ * h;
  float j10 = jc * e + jd * g;
  float j11 = jc * f + jd * h;
  pred_jac1[base]           = j00;
  pred_jac1[base + HW]      = j01;
  pred_jac1[base + 2L * HW] = j10;
  pred_jac1[base + 3L * HW] = j11;

  float hv = heat[((long)b * NKP + kp) * HW + pix];
  float vals[4] = {hv * j00, hv * j01, hv * j10, hv * j11};
  for (int q = 0; q < 4; ++q) {
    red[t] = vals[q]; __syncthreads();
    for (int s = 128; s > 0; s >>= 1) {
      if (t < s) red[t] += red[t + s];
      __syncthreads();
    }
    if (t == 0) atomicAdd(&jac1[((long)b * NKP + kp) * 4 + q], red[0]);
    __syncthreads();
  }
}

// ===========================================================================
// Host orchestration
// ===========================================================================
extern "C" void kernel_launch(void* const* d_in, const int* in_sizes, int n_in,
                              void* d_out, int out_size, void* d_ws, size_t ws_size,
                              hipStream_t stream) {
  (void)n_in; (void)out_size; (void)ws_size;
  const int H = HDIM, W = WDIM, HW = H * W;
  const int B = in_sizes[0] / (3 * HW);

  const float* frame[2] = {(const float*)d_in[0], (const float*)d_in[1]};
  const float* pv0 = (const float*)d_in[2];
  const float* pj0 = (const float*)d_in[3];
  const float* dw[5], *db[5], *uw[5], *ub[5];
  for (int i = 0; i < 5; ++i) { dw[i] = (const float*)d_in[4 + 2 * i];  db[i] = (const float*)d_in[5 + 2 * i]; }
  for (int i = 0; i < 5; ++i) { uw[i] = (const float*)d_in[14 + 2 * i]; ub[i] = (const float*)d_in[15 + 2 * i]; }
  const float* jw = (const float*)d_in[24];
  const float* jb = (const float*)d_in[25];

  const int dCinP[5]  = {32, 64, 128, 256, 512};
  const int dCinR[5]  = {3, 64, 128, 256, 512};
  const int dCout[5]  = {64, 128, 256, 512, 1024};
  const int uCinP[5]  = {1024, 576, 320, 192, 128};
  const int uC1[5]    = {1024, 64, 64, 64, 64};

  size_t off = 0;
  auto alloc = [&](size_t bytes) {
    off = (off + 255) & ~(size_t)255;
    size_t o = off; off += bytes; return o;
  };
  char* ws = (char*)d_ws;

  size_t o_pwD[5], o_pwU[5];
  for (int i = 0; i < 5; ++i)
    o_pwD[i] = alloc((size_t)9 * (dCinP[i] / 32) * dCout[i] * 32 * 2);
  for (int i = 0; i < 5; ++i)
    o_pwU[i] = alloc((size_t)9 * (uCinP[i] / 32) * 64 * 32 * 2);
  size_t o_pwJ = alloc((size_t)49 * (NDISPP / 32) * 64 * 32 * 2);

  size_t o_frm  = alloc((size_t)B * HW * 32 * 2);
  size_t o_tmpA = alloc((size_t)B * HW * 64 * 2);
  size_t o_down[5];
  for (int i = 0; i < 5; ++i) {
    int r = H >> (i + 1);
    o_down[i] = alloc((size_t)B * r * r * dCout[i] * 2);
  }
  size_t o_t[4];
  for (int i = 0; i < 4; ++i) {
    int r = 8 << i;
    o_t[i] = alloc((size_t)B * r * r * 64 * 2);
  }
  size_t o_fm0   = alloc((size_t)B * HW * FMC * 4);
  size_t o_fm1   = alloc((size_t)B * HW * FMC * 4);
  size_t o_means = alloc((size_t)B * FMC * 4);
  size_t o_corrF = alloc((size_t)B * HW * NDISP * 4);
  size_t o_corrH = alloc((size_t)B * HW * NDISPP * 2);
  size_t o_pj1   = alloc((size_t)B * 4 * NKP * HW * 4);
  size_t o_heat  = alloc((size_t)B * NKP * HW * 4);

  float* out = (float*)d_out;
  long q_jm  = 0;
  long q_val = (long)B * 4 * NKP * HW;
  long q_pv1 = q_val + (long)B * NKP * 2;
  long q_pj  = q_pv1 + (long)B * NKP * HW;
  long q_jac = q_pj + (long)B * 4 * NKP * HW;

  auto g1 = [](long n) { return dim3((unsigned)((n + 255) / 256)); };

  // ---- pack weights ----
  for (int i = 0; i < 5; ++i) {
    long tot = (long)9 * (dCinP[i] / 32) * dCout[i] * 32;
    pack_w_k<<<g1(tot), 256, 0, stream>>>(dw[i], (_Float16*)(ws + o_pwD[i]),
                                          dCinR[i], dCinP[i], dCout[i], dCout[i], 9);
  }
  for (int i = 0; i < 5; ++i) {
    long tot = (long)9 * (uCinP[i] / 32) * 64 * 32;
    pack_w_k<<<g1(tot), 256, 0, stream>>>(uw[i], (_Float16*)(ws + o_pwU[i]),
                                          uCinP[i], uCinP[i], 64, 64, 9);
  }
  {
    long tot = (long)49 * (NDISPP / 32) * 64 * 32;
    pack_w_k<<<g1(tot), 256, 0, stream>>>(jw, (_Float16*)(ws + o_pwJ),
                                          NDISP, NDISPP, 4 * NKP, 64, 49);
  }

  // conv launcher: 32-pixel M tiles, 64-channel N tiles, up to 8 waves/block
  auto conv = [&](const _Float16* inA, const _Float16* inB, int C1,
                  const _Float16* pw, const float* bias, int relu,
                  _Float16* outH, float* oNHWC, float* oNCHW,
                  int h, int w, int CinP, int CoutP, int CoutR,
                  int KH, int KW, int pad, int up) {
    int hw = h * w;
    int mt = hw / 32;
    int nwaves = (mt < 8) ? mt : 8;
    dim3 grid(mt / nwaves, CoutP / 64, B);
    conv_wmma_k<<<grid, nwaves * 32, 0, stream>>>(
        inA, inB, C1, pw, bias, relu, outH, oNHWC, oNCHW,
        h, w, CinP, CoutP, CoutR, KH, KW, pad, up);
  };

  // ---- feature extractor per frame ----
  for (int fidx = 0; fidx < 2; ++fidx) {
    float* fm = (float*)(ws + (fidx == 0 ? o_fm0 : o_fm1));
    pack_frame_k<<<g1((long)B * HW * 32), 256, 0, stream>>>(
        frame[fidx], (_Float16*)(ws + o_frm), B, HW);

    for (int i = 0; i < 5; ++i) {
      int r = H >> i;
      const _Float16* src = (i == 0) ? (const _Float16*)(ws + o_frm)
                                     : (const _Float16*)(ws + o_down[i - 1]);
      conv(src, src, dCinP[i], (const _Float16*)(ws + o_pwD[i]), db[i], 1,
           (_Float16*)(ws + o_tmpA), nullptr, nullptr,
           r, r, dCinP[i], dCout[i], dCout[i], 3, 3, 1, 0);
      int ro = r >> 1;
      pool2_k<<<g1((long)B * ro * ro * dCout[i]), 256, 0, stream>>>(
          (const _Float16*)(ws + o_tmpA), (_Float16*)(ws + o_down[i]),
          B, ro, ro, dCout[i]);
    }

    for (int i = 0; i < 5; ++i) {
      int r = 8 << i;
      const _Float16* inA = (i == 0) ? (const _Float16*)(ws + o_down[4])
                                     : (const _Float16*)(ws + o_t[i - 1]);
      const _Float16* inB = (i == 0) ? inA
                                     : (const _Float16*)(ws + o_down[4 - i]);
      _Float16* oh = (i < 4) ? (_Float16*)(ws + o_t[i]) : nullptr;
      float* of = (i == 4) ? fm : nullptr;
      conv(inA, inB, uC1[i], (const _Float16*)(ws + o_pwU[i]), ub[i], 1,
           oh, of, nullptr, r, r, uCinP[i], 64, 64, 3, 3, 1, 1);
    }

    fm_mean_k<<<dim3(B * FMC), 256, 0, stream>>>(fm, (float*)(ws + o_means), HW, FMC);
    fm_norm_k<<<g1((long)B * HW), 256, 0, stream>>>(fm, (float*)(ws + o_means), B, HW, FMC);
  }

  // ---- cost volume ----
  costvol_k<<<dim3(W, H, B), NDISPP, 0, stream>>>(
      (const float*)(ws + o_fm0), (const float*)(ws + o_fm1),
      (float*)(ws + o_corrF), (_Float16*)(ws + o_corrH), H, W);

  // ---- softmax over displacements + weighted aggregation ----
  smax_agg_k<<<dim3(W, H, B), 256, 0, stream>>>(
      (const float*)(ws + o_corrF), pv0, pj0,
      out + q_pv1, (float*)(ws + o_pj1), H, W);

  // ---- heatmap softmax + soft-argmax -> value ----
  heat_value_k<<<dim3(NKP, B), 256, 0, stream>>>(
      out + q_pv1, (float*)(ws + o_heat), out + q_val, H, W);

  // ---- jacobian head: 7x7 WMMA conv (169->40, padded 192->64) ----
  conv((const _Float16*)(ws + o_corrH), (const _Float16*)(ws + o_corrH), NDISPP,
       (const _Float16*)(ws + o_pwJ), jb, 0,
       nullptr, nullptr, out + q_jm,
       H, W, NDISPP, 64, 4 * NKP, 7, 7, 3, 0);

  // ---- 2x2 matmuls + heat-weighted reduction ----
  zero_f32_k<<<g1((long)B * 4 * NKP), 256, 0, stream>>>(out + q_jac, (long)B * 4 * NKP);
  jac_k<<<dim3(HW / 256, NKP, B), 256, 0, stream>>>(
      out + q_jm, (const float*)(ws + o_pj1), (const float*)(ws + o_heat),
      out + q_pj, out + q_jac, HW);
}
